// SAGE_9672266350774
// MI455X (gfx1250) — compile-verified, matrix operand order checked by reading, not run on hardware
//
#include <hip/hip_runtime.h>

typedef __bf16 bf16_t;
typedef __attribute__((ext_vector_type(16))) __bf16 v16bf;
typedef __attribute__((ext_vector_type(8)))  float   v8f;

#define FEAT 128

// ---------------------------------------------------------------------------
// WMMA 16x16x32 bf16 fragment index math (CDNA5 ISA 7.12.2, 16-bit A 16x32):
//   lanes 0-15  : row = lane,     VGPR v: K = (v<4?0:16) + 2*(v&3) + j
//   lanes 16-31 : row = lane-16,  same + 8
// B (32x16, KxN) mirrors with N in place of M; B[k][n] == W[n][k] for
// out = in @ W^T, so row-major W indexes directly as the B fragment.
// ---------------------------------------------------------------------------
__device__ __forceinline__ int frag_k(int i, int half) {
  const int v = i >> 1, j = i & 1;
  return ((v & 4) ? 16 : 0) + half * 8 + ((v & 3) << 1) + j;
}

__device__ __forceinline__ v16bf load_tile16(const bf16_t* __restrict__ base,
                                             int ld, int row, int k0, int half) {
  v16bf r;
#pragma unroll
  for (int i = 0; i < 16; ++i)
    r[i] = base[(size_t)row * ld + k0 + frag_k(i, half)];
  return r;
}

// ---------------------------------------------------------------------------
// Fused SAGE linear: out[m,:] = mean[m,:] @ Wl^T + x[m,:] @ Wr^T + b
// 8 waves/block, each wave owns 16 of the 128 output columns; weights stay in
// VGPRs. The 16x256 bf16 A-tile ([mean|x], 8 KB) is staged once per block in
// LDS via gfx1250 async global->LDS DMA, double-buffered so the DMA for tile
// t+1 overlaps the 8 WMMAs on tile t. All 8 A fragments are preloaded from
// LDS before the WMMA chain so ds_loads pipeline instead of wait_dscnt 0.
// ---------------------------------------------------------------------------
template <bool RELU, bool STORE_F, bool STORE_B>
__global__ __launch_bounds__(256)
void sage_gemm_wmma(const bf16_t* __restrict__ meanb, const bf16_t* __restrict__ xb,
                    const bf16_t* __restrict__ Wl, const bf16_t* __restrict__ Wr,
                    const float* __restrict__ bias,
                    float* __restrict__ out_f, bf16_t* __restrict__ out_b,
                    int n_nodes)
{
  __shared__ bf16_t tileA[2][16 * 2 * FEAT];   // 2 x 8KB double buffer

  const int lane = threadIdx.x & 31;
  const int wave = threadIdx.x >> 5;
  const int half = lane >> 4;
  const int l16  = lane & 15;
  const int n0   = wave * 16;                  // 8 waves cover 128 outputs

  v16bf Bf[8];
#pragma unroll
  for (int ks = 0; ks < 4; ++ks)
    Bf[ks]     = load_tile16(Wl, FEAT, n0 + l16, ks * 32, half);
#pragma unroll
  for (int ks = 0; ks < 4; ++ks)
    Bf[4 + ks] = load_tile16(Wr, FEAT, n0 + l16, ks * 32, half);

  const float bn = bias[n0 + l16];
  const int m_tiles = (n_nodes + 15) >> 4;

  // wave 0 DMAs one 16x256 bf16 tile (rows = [mean(128) | x(128)]) into LDS:
  // 512 x 16B chunks, 32 lanes x 16 issues of async b128.
  auto issue_tile = [&](int mt, int buf) {
    if (wave == 0) {
      const uint32_t lbase = (uint32_t)(size_t)(&tileA[buf][0]);
#pragma unroll
      for (int i = 0; i < 16; ++i) {
        const int c   = i * 32 + lane;         // chunk id 0..511
        const int row = c >> 5;                // 32 chunks per row
        const int cc  = c & 31;                // chunk within row
        int node = mt * 16 + row;
        if (node >= n_nodes) node = n_nodes - 1;
        const bf16_t* g = (cc < 16)
            ? meanb + (size_t)node * FEAT + cc * 8
            : xb    + (size_t)node * FEAT + (cc - 16) * 8;
        const uint32_t loff = lbase + (uint32_t)c * 16;
        asm volatile("global_load_async_to_lds_b128 %0, %1, off"
                     :: "v"(loff), "v"(g) : "memory");
      }
    }
  };

  const int mt0 = blockIdx.x;
  if (mt0 < m_tiles) issue_tile(mt0, 0);
  asm volatile("s_wait_asynccnt 0x0" ::: "memory");
  __syncthreads();

  int buf = 0;
  for (int mt = mt0; mt < m_tiles; mt += gridDim.x, buf ^= 1) {
    const int mtn = mt + gridDim.x;
    if (mtn < m_tiles) issue_tile(mtn, buf ^ 1);   // overlap DMA with compute

    // preload all 8 A fragments (16 x ds_load_b128, pipelined)
    v16bf Af[8];
#pragma unroll
    for (int ks = 0; ks < 8; ++ks)
#pragma unroll
      for (int i = 0; i < 16; ++i)
        Af[ks][i] = tileA[buf][l16 * (2 * FEAT) + ks * 32 + frag_k(i, half)];

    v8f acc = {};
#pragma unroll
    for (int ks = 0; ks < 8; ++ks)
      acc = __builtin_amdgcn_wmma_f32_16x16x32_bf16(false, Af[ks], false, Bf[ks],
                                                    (short)0, acc, false, false);

    const int m0 = mt << 4;
#pragma unroll
    for (int r = 0; r < 8; ++r) {            // D layout: VGPR r -> M = r (+8 hi half)
      const int m = m0 + r + half * 8;
      if (m >= n_nodes) continue;
      float v = acc[r] + bn;
      if constexpr (RELU) v = v > 0.f ? v : 0.f;
      const size_t idx = (size_t)m * FEAT + n0 + l16;
      if constexpr (STORE_F) out_f[idx] = v;
      if constexpr (STORE_B) out_b[idx] = (bf16_t)v;
    }

    asm volatile("s_wait_asynccnt 0x0" ::: "memory");
    __syncthreads();                          // next tile's LDS buffer is ready
  }
}

// ---------------------------------------------------------------------------
// CSR build: degree -> exclusive scan -> bucket fill. Replaces 205M fp32
// atomics/layer with a one-time 1.6M int-atomic build + pure gathers.
// ---------------------------------------------------------------------------
__global__ void degree_count(const int* __restrict__ dst, int* __restrict__ cnt,
                             int nedges) {
  const int t = blockIdx.x * blockDim.x + threadIdx.x;
  if (t < nedges) atomicAdd(&cnt[dst[t]], 1);
}

__global__ void block_sums(const int* __restrict__ cnt, int* __restrict__ bsum, int n) {
  __shared__ int sh[256];
  const int i = blockIdx.x * 256 + threadIdx.x;
  sh[threadIdx.x] = (i < n) ? cnt[i] : 0;
  __syncthreads();
  for (int s = 128; s > 0; s >>= 1) {
    if (threadIdx.x < s) sh[threadIdx.x] += sh[threadIdx.x + s];
    __syncthreads();
  }
  if (threadIdx.x == 0) bsum[blockIdx.x] = sh[0];
}

__global__ void scan_bsum_serial(int* __restrict__ bsum, int nb) {
  int run = 0;
  for (int b = 0; b < nb; ++b) { const int t = bsum[b]; bsum[b] = run; run += t; }
}

__global__ void row_scan(const int* __restrict__ cnt, const int* __restrict__ bofs,
                         int* __restrict__ rs, int n) {
  __shared__ int sh[256];
  const int i = blockIdx.x * 256 + threadIdx.x;
  const int v = (i < n) ? cnt[i] : 0;
  sh[threadIdx.x] = v;
  __syncthreads();
  for (int off = 1; off < 256; off <<= 1) {
    const int t = (threadIdx.x >= off) ? sh[threadIdx.x - off] : 0;
    __syncthreads();
    sh[threadIdx.x] += t;
    __syncthreads();
  }
  const int excl = bofs[blockIdx.x] + sh[threadIdx.x] - v;
  if (i < n) rs[i] = excl;
  if (i == n - 1) rs[n] = excl + v;
}

__global__ void csr_fill(const int* __restrict__ src, const int* __restrict__ dst,
                         const int* __restrict__ rs, int* __restrict__ fill,
                         int* __restrict__ csr_src, int nedges) {
  const int t = blockIdx.x * blockDim.x + threadIdx.x;
  if (t >= nedges) return;
  const int d = dst[t];
  const int p = atomicAdd(&fill[d], 1);
  csr_src[rs[d] + p] = src[t];
}

// ---------------------------------------------------------------------------
// Gather-style mean aggregation: one wave per node, 4 feats/lane, in-register
// accumulation (zero atomics; x/h are L2-resident on the 192MB L2). Two edges
// in flight per iteration to double memory-level parallelism of the random
// L2 gathers (avg degree 16 => otherwise a serial dependent-load chain).
// ---------------------------------------------------------------------------
__global__ void gather_mean_f32(const float* __restrict__ x, const int* __restrict__ csr,
                                const int* __restrict__ rs, bf16_t* __restrict__ mean,
                                int n_nodes) {
  const long long t = (long long)blockIdx.x * blockDim.x + threadIdx.x;
  const int w = (int)(t >> 5);
  if (w >= n_nodes) return;
  const int lane = (int)(t & 31);
  const int beg = rs[w], end = rs[w + 1];
  float a0 = 0.f, a1 = 0.f, a2 = 0.f, a3 = 0.f;
  float c0 = 0.f, c1 = 0.f, c2 = 0.f, c3 = 0.f;
  int j = beg;
  for (; j + 1 < end; j += 2) {
    if (j + 4 < end)
      __builtin_prefetch(x + (size_t)csr[j + 4] * FEAT + lane * 4, 0, 0);
    const float4 v0 = *(const float4*)(x + (size_t)csr[j]     * FEAT + lane * 4);
    const float4 v1 = *(const float4*)(x + (size_t)csr[j + 1] * FEAT + lane * 4);
    a0 += v0.x; a1 += v0.y; a2 += v0.z; a3 += v0.w;
    c0 += v1.x; c1 += v1.y; c2 += v1.z; c3 += v1.w;
  }
  if (j < end) {
    const float4 v0 = *(const float4*)(x + (size_t)csr[j] * FEAT + lane * 4);
    a0 += v0.x; a1 += v0.y; a2 += v0.z; a3 += v0.w;
  }
  const float iv = 1.0f / fmaxf((float)(end - beg), 1.0f);
  const size_t o = (size_t)w * FEAT + lane * 4;
  mean[o + 0] = (bf16_t)((a0 + c0) * iv);
  mean[o + 1] = (bf16_t)((a1 + c1) * iv);
  mean[o + 2] = (bf16_t)((a2 + c2) * iv);
  mean[o + 3] = (bf16_t)((a3 + c3) * iv);
}

__global__ void gather_mean_bf16(const bf16_t* __restrict__ h, const int* __restrict__ csr,
                                 const int* __restrict__ rs, bf16_t* __restrict__ mean,
                                 int n_nodes) {
  const long long t = (long long)blockIdx.x * blockDim.x + threadIdx.x;
  const int w = (int)(t >> 5);
  if (w >= n_nodes) return;
  const int lane = (int)(t & 31);
  const int beg = rs[w], end = rs[w + 1];
  float a0 = 0.f, a1 = 0.f, a2 = 0.f, a3 = 0.f;
  float c0 = 0.f, c1 = 0.f, c2 = 0.f, c3 = 0.f;
  int j = beg;
  for (; j + 1 < end; j += 2) {
    if (j + 4 < end)
      __builtin_prefetch(h + (size_t)csr[j + 4] * FEAT + lane * 4, 0, 0);
    const bf16_t* p0 = h + (size_t)csr[j]     * FEAT + lane * 4;
    const bf16_t* p1 = h + (size_t)csr[j + 1] * FEAT + lane * 4;
    a0 += (float)p0[0]; a1 += (float)p0[1]; a2 += (float)p0[2]; a3 += (float)p0[3];
    c0 += (float)p1[0]; c1 += (float)p1[1]; c2 += (float)p1[2]; c3 += (float)p1[3];
  }
  if (j < end) {
    const bf16_t* p0 = h + (size_t)csr[j] * FEAT + lane * 4;
    a0 += (float)p0[0]; a1 += (float)p0[1]; a2 += (float)p0[2]; a3 += (float)p0[3];
  }
  const float iv = 1.0f / fmaxf((float)(end - beg), 1.0f);
  const size_t o = (size_t)w * FEAT + lane * 4;
  mean[o + 0] = (bf16_t)((a0 + c0) * iv);
  mean[o + 1] = (bf16_t)((a1 + c1) * iv);
  mean[o + 2] = (bf16_t)((a2 + c2) * iv);
  mean[o + 3] = (bf16_t)((a3 + c3) * iv);
}

__global__ void cvt_bf16(const float* __restrict__ in, bf16_t* __restrict__ out, int n) {
  const int t = blockIdx.x * blockDim.x + threadIdx.x;
  if (t < n) out[t] = (bf16_t)in[t];
}

// ---------------------------------------------------------------------------
extern "C" void kernel_launch(void* const* d_in, const int* in_sizes, int n_in,
                              void* d_out, int out_size, void* d_ws, size_t ws_size,
                              hipStream_t stream) {
  const float* x   = (const float*)d_in[0];
  const int*   ei  = (const int*)d_in[1];
  const float* W1l = (const float*)d_in[2];
  const float* W1r = (const float*)d_in[3];
  const float* b1  = (const float*)d_in[4];
  const float* W2l = (const float*)d_in[5];
  const float* W2r = (const float*)d_in[6];
  const float* b2  = (const float*)d_in[7];

  const int N = in_sizes[0] / FEAT;
  const int E = in_sizes[1] / 2;
  const int* src = ei;
  const int* dst = ei + E;

  const int TB = 256;
  const int NF = N * FEAT;
  const int WW = FEAT * FEAT;
  const int nb = (N + TB - 1) / TB;

  // workspace carve-out (~85 MB)
  char* ws = (char*)d_ws;
  auto carve = [&](size_t bytes) -> char* {
    char* p = ws;
    ws += (bytes + 255) & ~(size_t)255;
    return p;
  };
  int*    cnt_i = (int*)carve((size_t)N * 4);
  int*    fill  = (int*)carve((size_t)N * 4);
  int*    bsum  = (int*)carve((size_t)nb * 4);
  int*    rs    = (int*)carve((size_t)(N + 1) * 4);
  int*    csr   = (int*)carve((size_t)E * 4);
  bf16_t* meanb = (bf16_t*)carve((size_t)NF * 2);
  bf16_t* xb    = (bf16_t*)carve((size_t)NF * 2);
  bf16_t* hb    = (bf16_t*)carve((size_t)NF * 2);
  bf16_t* wb    = (bf16_t*)carve((size_t)4 * WW * 2);
  bf16_t *W1lb = wb, *W1rb = wb + WW, *W2lb = wb + 2 * WW, *W2rb = wb + 3 * WW;

  hipMemsetAsync(cnt_i, 0, (size_t)N * 4, stream);
  hipMemsetAsync(fill,  0, (size_t)N * 4, stream);

  cvt_bf16<<<(NF + TB - 1) / TB, TB, 0, stream>>>(x, xb, NF);
  cvt_bf16<<<(WW + TB - 1) / TB, TB, 0, stream>>>(W1l, W1lb, WW);
  cvt_bf16<<<(WW + TB - 1) / TB, TB, 0, stream>>>(W1r, W1rb, WW);
  cvt_bf16<<<(WW + TB - 1) / TB, TB, 0, stream>>>(W2l, W2lb, WW);
  cvt_bf16<<<(WW + TB - 1) / TB, TB, 0, stream>>>(W2r, W2rb, WW);

  // CSR build (reused by both layers)
  degree_count<<<(E + TB - 1) / TB, TB, 0, stream>>>(dst, cnt_i, E);
  block_sums<<<nb, TB, 0, stream>>>(cnt_i, bsum, N);
  scan_bsum_serial<<<1, 1, 0, stream>>>(bsum, nb);
  row_scan<<<nb, TB, 0, stream>>>(cnt_i, bsum, rs, N);
  csr_fill<<<(E + TB - 1) / TB, TB, 0, stream>>>(src, dst, rs, fill, csr, E);

  const int ga_blocks = (int)(((long long)N * 32 + TB - 1) / TB);
  const int m_tiles = (N + 15) / 16;
  const int g_blocks = m_tiles < 512 ? m_tiles : 512;

  // ---- layer 1 ----
  gather_mean_f32<<<ga_blocks, TB, 0, stream>>>(x, csr, rs, meanb, N);
  sage_gemm_wmma<true, false, true><<<g_blocks, TB, 0, stream>>>(
      meanb, xb, W1lb, W1rb, b1, nullptr, hb, N);

  // ---- layer 2 ----
  gather_mean_bf16<<<ga_blocks, TB, 0, stream>>>(hb, csr, rs, meanb, N);
  sage_gemm_wmma<false, true, false><<<g_blocks, TB, 0, stream>>>(
      meanb, hb, W2lb, W2rb, b2, (float*)d_out, nullptr, N);
}
